// GATr_14645838479726
// MI455X (gfx1250) — matrix operand order repaired and checked by reading, not compiled, not run on hardware
//
#include <hip/hip_runtime.h>
#include <hip/hip_bf16.h>
#include <math.h>
#include <stdint.h>

// ---------------------------------------------------------------------------
// Types / helpers
// ---------------------------------------------------------------------------
typedef unsigned short u16;
typedef __attribute__((ext_vector_type(16))) __bf16 v16bf;
typedef __attribute__((ext_vector_type(8)))  float  v8f;

#if defined(__has_builtin)
#if __has_builtin(__builtin_amdgcn_global_load_async_to_lds_b128)
#define USE_ASYNC_LDS 1
#endif
#endif

#ifdef USE_ASYNC_LDS
// Builtin signature (from clang diagnostic): params are pointers to
// int __attribute__((vector_size(16))) in AS1 (global) / AS3 (LDS).
typedef int v4i32 __attribute__((vector_size(16)));
typedef v4i32 __attribute__((address_space(1))) * g_v4ptr;
typedef v4i32 __attribute__((address_space(3))) * l_v4ptr;
static __device__ inline void async_b128(const void* g, void* l) {
    // generic->AS1 / AS3 via integer round-trip (low 32 bits of a generic LDS
    // pointer are the raw LDS offset; AS3 pointers are 32-bit).
    __builtin_amdgcn_global_load_async_to_lds_b128(
        (g_v4ptr)(unsigned long long)(uintptr_t)g,
        (l_v4ptr)(unsigned int)(uintptr_t)l, 0, 0);
}
static __device__ inline void async_wait0() {
    asm volatile("s_wait_asynccnt 0x0" ::: "memory");
}
#endif

static __device__ inline u16 f2bf(float f) {
    unsigned int u = __float_as_uint(f);
    unsigned int r = u + 0x7FFFu + ((u >> 16) & 1u);
    return (u16)(r >> 16);
}

static __device__ inline v16bf ld16x2(const u16* p0, const u16* p1) {
    union { v16bf v; uint4 q[2]; } r;
    r.q[0] = *(const uint4*)p0;
    r.q[1] = *(const uint4*)p1;
    return r.v;
}

static __device__ inline v8f wmma_bf16(v16bf a, v16bf b, v8f c) {
    return __builtin_amdgcn_wmma_f32_16x16x32_bf16(false, a, false, b, (short)0, c,
                                                   false, false);
}

static __device__ inline float gelu_exact(float x) {
    return 0.5f * x * (1.0f + erff(x * 0.70710678118654752f));
}

// Blade tables: order [1,e0,e1,e2,e3,e01,e02,e03,e12,e13,e23,e012,e013,e023,e123,e0123]
__device__ static const int c_grade[16] = {0,1,1,1,1,2,2,2,2,2,2,3,3,3,3,4};
__device__ static const int c_e0idx[16] = {1,-1,5,6,7,-1,-1,-1,11,12,13,-1,-1,-1,15,-1};
__device__ static const int c_mask[16]  = {0,1,2,4,8,3,5,9,6,10,12,7,11,13,14,15};
__device__ static const int c_idxof[16] = {0,1,2,5,3,6,8,11,4,7,9,12,10,13,14,15};

// ---------------------------------------------------------------------------
// Table builder: GP and JOIN 16x16x16 tables derived on device.
// ---------------------------------------------------------------------------
__global__ __launch_bounds__(256) void build_tables(float* GPt, float* JNt) {
    __shared__ float gp[16][16][16];
    __shared__ float wg[16][16][16];
    __shared__ int   dualT[16];
    __shared__ float dualS[16];
    int tid = threadIdx.x;
    for (int i = tid; i < 4096; i += 256) { ((float*)gp)[i] = 0.f; ((float*)wg)[i] = 0.f; }
    __syncthreads();
    {
        int i = tid >> 4, j = tid & 15;
        int a = c_mask[i], b = c_mask[j];
        float coeff = 1.f; int res = a;               // geometric product (e0^2 = 0)
        for (int e = 0; e < 4; e++) if ((b >> e) & 1) {
            if (__popc(res >> (e + 1)) & 1) coeff = -coeff;
            if ((res >> e) & 1) { if (e == 0) coeff = 0.f; res &= ~(1 << e); }
            else res |= (1 << e);
        }
        if (coeff != 0.f) gp[c_idxof[res]][i][j] = coeff;
        if ((a & b) == 0) {                           // wedge
            float s = 1.f; int r2 = a;
            for (int e = 0; e < 4; e++) if ((b >> e) & 1) {
                if (__popc(r2 >> (e + 1)) & 1) s = -s;
                r2 |= (1 << e);
            }
            wg[c_idxof[r2]][i][j] = s;
        }
    }
    if (tid < 16) {
        int a = c_mask[tid], comp = (~a) & 15, inv = 0;
        for (int x = 0; x < 4; x++) if ((a >> x) & 1) inv += __popc(comp & ((1 << x) - 1));
        dualT[tid] = c_idxof[comp];
        dualS[tid] = (inv & 1) ? -1.f : 1.f;
    }
    __syncthreads();
    for (int i = tid; i < 4096; i += 256) {
        int k = i >> 8, b = (i >> 4) & 15, c = i & 15;
        GPt[i] = gp[k][b][c];
        JNt[i] = dualS[k] * dualS[b] * dualS[c] * wg[dualT[k]][dualT[b]][dualT[c]];
    }
}

// ---------------------------------------------------------------------------
// ref_scale: mean over (S, C) of mv component e0123 per batch
// ---------------------------------------------------------------------------
__global__ __launch_bounds__(256) void refscale_k(const float* mv, float* refs, int S) {
    int b = blockIdx.x, tid = threadIdx.x;
    float acc = 0.f;
    for (int i = tid; i < S * 4; i += 256)
        acc += mv[((long)b * S * 4 + i) * 16 + 15];
    __shared__ float red[256];
    red[tid] = acc; __syncthreads();
    for (int s = 128; s > 0; s >>= 1) { if (tid < s) red[tid] += red[tid + s]; __syncthreads(); }
    if (tid == 0) refs[b] = red[0] / (S * 4.0f);
}

// ---------------------------------------------------------------------------
// Expand equi_linear params into TRANSPOSED dense bf16 weights Wt[Mp][Kp]
// (transposed so the GEMM's B-operand staging and fragment reads are all
// contiguous along K -> vector ds/global ops, no scalar transpose in LDS).
// Logical rows (K): cin*16 mv inputs, sin scalar inputs, 1 bias row, pad.
// Logical cols (M): cout*16 mv outputs, sout scalar outputs, pad.
// ---------------------------------------------------------------------------
__global__ __launch_bounds__(256) void prep_w(
    const float* w_mv, const float* b_mv, const float* w_s2mv,
    const float* w_s, const float* w_mv2s, const float* b_s,
    int cin, int cout, int sin, int sout, int Kp, int Mp, u16* Wt) {
    int idx = blockIdx.x * 256 + threadIdx.x;
    if (idx >= Kp * Mp) return;
    int col = idx / Kp, row = idx % Kp;          // idx = col*Kp + row (transposed)
    int K = cin * 16 + sin + 1, M = cout * 16 + sout;
    float v = 0.f;
    if (row < K && col < M) {
        if (col < cout * 16) {
            int o = col >> 4, j = col & 15;
            if (row < cin * 16) {
                int i = row >> 4, k = row & 15;
                const float* wp = w_mv + (long)(o * cin + i) * 9;
                if (j == k) v += wp[c_grade[k]];
                if (c_e0idx[k] == j) v += wp[5 + c_grade[k]];
            } else if (row < cin * 16 + sin) {
                if (j == 0 && w_s2mv) v = w_s2mv[(long)o * sin + (row - cin * 16)];
            } else {
                if (j == 0) v = b_mv[o];
            }
        } else {
            int so = col - cout * 16;
            if (row < cin * 16) {
                if ((row & 15) == 0 && w_mv2s) v = w_mv2s[(long)so * cin + (row >> 4)];
            } else if (row < cin * 16 + sin) {
                if (w_s) v = w_s[(long)so * sin + (row - cin * 16)];
            } else {
                if (b_s) v = b_s[so];
            }
        }
    }
    Wt[idx] = f2bf(v);
}

// ---------------------------------------------------------------------------
// Pack: optional equi_norm + f32 -> bf16 + bias column. One wave per token.
// ---------------------------------------------------------------------------
__global__ __launch_bounds__(256) void pack_k(
    const float* mvp, int mvs, const float* sp, int ss,
    int cin, int sin, int Kp, int donorm, u16* out, int N) {
    int gid = blockIdx.x * 256 + threadIdx.x;
    int t = gid >> 5, lane = gid & 31;
    if (t >= N) return;
    const float* mr = mvp + (long)t * mvs;
    const float* sr = sp + (long)t * ss;
    float msc = 1.f, ssc = 1.f;
    if (donorm) {
        float a = 0.f;
        for (int i = lane; i < cin * 16; i += 32)
            if ((0x471D >> (i & 15)) & 1) { float v = mr[i]; a += v * v; }
        for (int m = 16; m; m >>= 1) a += __shfl_xor(a, m);
        msc = rsqrtf(a / (float)cin + 0.01f);
        float b = 0.f;
        for (int i = lane; i < sin; i += 32) { float v = sr[i]; b += v * v; }
        for (int m = 16; m; m >>= 1) b += __shfl_xor(b, m);
        ssc = rsqrtf(b / (float)sin + 0.01f);
    }
    int K = cin * 16 + sin;
    u16* o = out + (long)t * Kp;
    for (int i = lane; i < Kp; i += 32) {
        float v;
        if (i < cin * 16)      v = mr[i] * msc;
        else if (i < K)        v = sr[i - cin * 16] * ssc;
        else if (i == K)       v = 1.f;
        else                   v = 0.f;
        o[i] = f2bf(v);
    }
}

// ---------------------------------------------------------------------------
// bf16 WMMA GEMM: Out[N x Mp] = X[N x Kp] @ Wt^T (Wt stored [Mp][Kp]).
// 64x64 tile per WG (8 waves), k-step 32. Async global->LDS staging when
// the toolchain exposes GLOBAL_LOAD_ASYNC_TO_LDS (ASYNCcnt tracked).
// ---------------------------------------------------------------------------
__global__ __launch_bounds__(256) void gemm_bf16(
    const u16* __restrict__ X, const u16* __restrict__ Wt,
    float* __restrict__ Out, int N, int Kp, int Mp) {
    __shared__ u16 As[64 * 32];   // [row][k]
    __shared__ u16 Bs[64 * 32];   // [n][k]
    int tid = threadIdx.x, wave = tid >> 5, lane = tid & 31;
    int nb = blockIdx.x * 64, mb = blockIdx.y * 64;
    int wm = (wave & 1) * 32, wn = (wave >> 1) * 16;
    int r = tid >> 2, c = (tid & 3) * 8;          // 16B per thread per tile
    v8f acc0 = {}, acc1 = {};
    for (int k0 = 0; k0 < Kp; k0 += 32) {
        const u16* srcA = X  + (long)(mb + r) * Kp + k0 + c;
        const u16* srcB = Wt + (long)(nb + r) * Kp + k0 + c;
#ifdef USE_ASYNC_LDS
        async_b128(srcA, &As[r * 32 + c]);
        async_b128(srcB, &Bs[r * 32 + c]);
#else
        *(uint4*)&As[r * 32 + c] = *(const uint4*)srcA;
        *(uint4*)&Bs[r * 32 + c] = *(const uint4*)srcB;
#endif
        if (k0 + 32 < Kp) {
            __builtin_prefetch(srcA + 32, 0, 3);
            __builtin_prefetch(srcB + 32, 0, 3);
        }
#ifdef USE_ASYNC_LDS
        async_wait0();
#endif
        __syncthreads();
        int arow = lane & 15;
        int kb = (lane >> 4) * 8;                 // A frag: K = kb..kb+7, kb+16..kb+23
        v16bf a0 = ld16x2(&As[(wm + arow) * 32 + kb],      &As[(wm + arow) * 32 + kb + 16]);
        v16bf a1 = ld16x2(&As[(wm + 16 + arow) * 32 + kb], &As[(wm + 16 + arow) * 32 + kb + 16]);
        const u16* bp = &Bs[(wn + (lane & 15)) * 32 + (lane >> 4) * 8];
        v16bf b0 = ld16x2(bp, bp + 16);
        acc0 = wmma_bf16(a0, b0, acc0);
        acc1 = wmma_bf16(a1, b0, acc1);
        __syncthreads();
    }
    int mrow = mb + wm + (lane >> 4) * 8;
    int ncol = nb + wn + (lane & 15);
    #pragma unroll
    for (int rr = 0; rr < 8; rr++) {
        Out[(long)(mrow + rr) * Mp + ncol]      = acc0[rr];
        Out[(long)(mrow + 16 + rr) * Mp + ncol] = acc1[rr];
    }
}

// ---------------------------------------------------------------------------
// Build per-head packed Q/K/V features (bf16).
// Qf/Kf: [B*H][S][32]   = 16 inner-mv dims + 8 scalars + 8 pad (Q pre-scaled)
// Vf:    [B*H][S/16][48][16]  (per-16-key transposed tiles: [dim][key])
// ---------------------------------------------------------------------------
__global__ __launch_bounds__(256) void qkvprep_k(
    const float* Q, const float* K, const float* V,
    u16* Qf, u16* Kf, u16* Vf, int N) {
    int gid = blockIdx.x * 256 + threadIdx.x;
    if (gid >= N * 8) return;
    int t = gid >> 3, h = gid & 7;
    const int IN[8] = {0, 2, 3, 4, 8, 9, 10, 14};
    long bh = (long)(t >> 10) * 8 + h;
    int srow = t & 1023;
    long row = bh * 1024 + srow;
    const float qs = 0.20412414523193154f; // 1/sqrt(8*hm + hs) = 1/sqrt(24)
    u16* qo = Qf + row * 32;
    u16* ko = Kf + row * 32;
    u16* vo = Vf + (bh * 64 + (srow >> 4)) * 768 + (srow & 15);
    const float* qr = Q + (long)t * 320;
    const float* kr = K + (long)t * 320;
    const float* vr = V + (long)t * 320;
    #pragma unroll
    for (int e = 0; e < 8; e++) {
        qo[e]      = f2bf(qr[(2 * h) * 16 + IN[e]] * qs);
        qo[8 + e]  = f2bf(qr[(2 * h + 1) * 16 + IN[e]] * qs);
        qo[16 + e] = f2bf(qr[256 + h * 8 + e] * qs);
        qo[24 + e] = 0;
        ko[e]      = f2bf(kr[(2 * h) * 16 + IN[e]]);
        ko[8 + e]  = f2bf(kr[(2 * h + 1) * 16 + IN[e]]);
        ko[16 + e] = f2bf(kr[256 + h * 8 + e]);
        ko[24 + e] = 0;
    }
    #pragma unroll
    for (int d = 0; d < 32; d++) vo[d * 16] = f2bf(vr[h * 32 + d]);
    #pragma unroll
    for (int e = 0; e < 8; e++) { vo[(32 + e) * 16] = f2bf(vr[256 + h * 8 + e]); vo[(40 + e) * 16] = 0; }
}

// ---------------------------------------------------------------------------
// Flash attention: one wave per 16-query tile, 4 waves/WG share K/V LDS tiles.
// Logits = WMMA over 32-dim padded feature; online softmax on C-frag rows;
// P recast C-layout -> A-layout through LDS; 3 WMMAs vs 48-dim value tile.
// Output written into flat [N][320] layout (heads concatenated).
// ---------------------------------------------------------------------------
__global__ __launch_bounds__(128) void attn_k(
    const u16* __restrict__ Qf, const u16* __restrict__ Kf,
    const u16* __restrict__ Vf, float* __restrict__ O, int S) {
    __shared__ u16 Kt[16 * 32];      // [key][feat]
    __shared__ u16 Vt[48 * 32];      // [dim][keyslot] (slots 16..31 = 0)
    __shared__ u16 Pst[4][16 * 32];  // per-wave probability staging (cols 16..31 = 0)
    int tid = threadIdx.x, wave = tid >> 5, lane = tid & 31;
    int bh = blockIdx.x >> 4, qg = blockIdx.x & 15;
    int qtile = qg * 64 + wave * 16;
    for (int i = tid; i < 48 * 16; i += 128) { int d = i / 16, k = i % 16; Vt[d * 32 + 16 + k] = 0; }
    for (int i = lane; i < 16 * 16; i += 32) { int r = i >> 4, c = i & 15; Pst[wave][r * 32 + 16 + c] = 0; }
    __syncthreads();

    int arow = lane & 15, kb = (lane >> 4) * 16;
    const u16* qbase = Qf + ((long)bh * S + qtile) * 32;
    v16bf aq = ld16x2(qbase + arow * 32 + kb, qbase + arow * 32 + kb + 8);

    float m[8], l[8];
    #pragma unroll
    for (int r2 = 0; r2 < 8; r2++) { m[r2] = -1e30f; l[r2] = 0.f; }
    v8f o0 = {}, o1 = {}, o2 = {};

    for (int kt = 0; kt < S; kt += 16) {
        __syncthreads();
        const u16* ksrc = Kf + ((long)bh * S + kt) * 32;                 // 16x32
        const u16* vsrc = Vf + ((long)bh * 64 + (kt >> 4)) * 768;        // 48x16
#ifdef USE_ASYNC_LDS
        if (tid < 64)  async_b128(ksrc + tid * 8, &Kt[tid * 8]);
        if (tid < 96)  async_b128(vsrc + tid * 8, &Vt[(tid >> 1) * 32 + (tid & 1) * 8]);
        async_wait0();
#else
        {   // K tile: 16 keys x 32 feats
            int key = tid >> 3, offk = (tid & 7) * 4;
            *(uint2*)&Kt[key * 32 + offk] = *(const uint2*)(ksrc + key * 32 + offk);
        }
        for (int i = tid; i < 192; i += 128)     // V tile (already transposed)
            *(uint2*)&Vt[(i >> 2) * 32 + (i & 3) * 4] = *(const uint2*)(vsrc + i * 4);
#endif
        __syncthreads();

        const u16* bp = &Kt[(lane & 15) * 32 + (lane >> 4) * 16];
        v16bf bk = ld16x2(bp, bp + 8);
        v8f s = {};
        s = wmma_bf16(aq, bk, s);

        #pragma unroll
        for (int r2 = 0; r2 < 8; r2++) {
            float x = s[r2];
            float rm = x;
            rm = fmaxf(rm, __shfl_xor(rm, 1));
            rm = fmaxf(rm, __shfl_xor(rm, 2));
            rm = fmaxf(rm, __shfl_xor(rm, 4));
            rm = fmaxf(rm, __shfl_xor(rm, 8));
            float mn = fmaxf(m[r2], rm);
            float sc = __expf(m[r2] - mn);
            float p  = __expf(x - mn);
            float rs = p;
            rs += __shfl_xor(rs, 1);
            rs += __shfl_xor(rs, 2);
            rs += __shfl_xor(rs, 4);
            rs += __shfl_xor(rs, 8);
            l[r2] = l[r2] * sc + rs;
            m[r2] = mn;
            o0[r2] *= sc; o1[r2] *= sc; o2[r2] *= sc;
            Pst[wave][(r2 + 8 * (lane >> 4)) * 32 + (lane & 15)] = f2bf(p);
        }
        v16bf ap = ld16x2(&Pst[wave][arow * 32 + kb], &Pst[wave][arow * 32 + kb + 8]);
        const u16* v0 = &Vt[(0 * 16 + (lane & 15)) * 32 + (lane >> 4) * 16];
        const u16* v1 = &Vt[(1 * 16 + (lane & 15)) * 32 + (lane >> 4) * 16];
        const u16* v2 = &Vt[(2 * 16 + (lane & 15)) * 32 + (lane >> 4) * 16];
        o0 = wmma_bf16(ap, ld16x2(v0, v0 + 8), o0);
        o1 = wmma_bf16(ap, ld16x2(v1, v1 + 8), o1);
        o2 = wmma_bf16(ap, ld16x2(v2, v2 + 8), o2);
    }

    int h = bh & 7;
    long tq0 = (long)(bh >> 3) * S + qtile;
    int n = lane & 15;
    #pragma unroll
    for (int r2 = 0; r2 < 8; r2++) {
        float inv = 1.f / l[r2];
        long t = tq0 + r2 + 8 * (lane >> 4);
        O[t * 320 + h * 32 + n]      = o0[r2] * inv;
        O[t * 320 + h * 32 + 16 + n] = o1[r2] * inv;
        if (n < 8) O[t * 320 + 256 + h * 8 + n] = o2[r2] * inv;
    }
}

// ---------------------------------------------------------------------------
// Geometric bilinears (GP for ch<16, JOIN*ref_scale for ch>=16) + scalar gate.
// ---------------------------------------------------------------------------
__global__ __launch_bounds__(256) void bilinear_k(
    const float* __restrict__ L, const float* __restrict__ R,
    const float* __restrict__ GPt, const float* __restrict__ JNt,
    const float* __restrict__ refs, float* __restrict__ O, int N) {
    __shared__ float sT[2][4096];
    int tid = threadIdx.x;
    for (int i = tid; i < 4096; i += 256) { sT[0][i] = GPt[i]; sT[1][i] = JNt[i]; }
    __syncthreads();
    int gid = blockIdx.x * 256 + tid;
    int t = gid >> 5, ch = gid & 31;
    if (t >= N) return;
    float lv[16], rv[16];
    const float* lp = L + (long)t * 640 + ch * 16;
    const float* rp = R + (long)t * 512 + ch * 16;
    #pragma unroll
    for (int i = 0; i < 16; i++) { lv[i] = lp[i]; rv[i] = rp[i]; }
    const float* Tb = (ch < 16) ? sT[0] : sT[1];
    float scale = (ch < 16) ? 1.f : refs[t >> 10];
    float out[16];
    for (int k = 0; k < 16; k++) {
        float acc = 0.f;
        for (int i = 0; i < 16; i++) {
            float li = lv[i];
            const float* tr = Tb + k * 256 + i * 16;
            #pragma unroll
            for (int j = 0; j < 16; j++) acc += tr[j] * rv[j] * li;
        }
        out[k] = acc * scale;
    }
    float g = gelu_exact(out[0]);
    float* op = O + (long)t * 640 + ch * 16;
    #pragma unroll
    for (int k = 0; k < 16; k++) op[k] = out[k] * g;
}

__global__ __launch_bounds__(256) void sgelu_k(const float* L, float* O, int N) {
    int idx = blockIdx.x * 256 + threadIdx.x;
    if (idx >= N * 128) return;
    int t = idx / 128, c = idx % 128;
    O[(long)t * 640 + 512 + c] = gelu_exact(L[(long)t * 640 + 512 + c]);
}

__global__ __launch_bounds__(256) void add_k(float* X, const float* T, int n) {
    int i = blockIdx.x * 256 + threadIdx.x;
    if (i < n) X[i] += T[i];
}

__global__ __launch_bounds__(256) void unpack_k(const float* G, float* out, int N) {
    int idx = blockIdx.x * 256 + threadIdx.x;
    if (idx >= N * 80) return;
    int t = idx / 80, c = idx % 80;
    if (c < 64) out[(long)t * 64 + c] = G[(long)t * 128 + c];
    else        out[(long)N * 64 + (long)t * 16 + (c - 64)] = G[(long)t * 128 + c];
}

// ---------------------------------------------------------------------------
// Host orchestration
// ---------------------------------------------------------------------------
extern "C" void kernel_launch(void* const* d_in, const int* in_sizes, int n_in,
                              void* d_out, int out_size, void* d_ws, size_t ws_size,
                              hipStream_t stream) {
    (void)in_sizes; (void)n_in; (void)out_size; (void)ws_size;
    const int N = 4096, Bz = 4, S = 1024, H = 8;

    char* wsc = (char*)d_ws;
    size_t off = 0;
    auto take = [&](size_t bytes) -> void* {
        void* p = wsc + off;
        off = (off + bytes + 255) & ~(size_t)255;
        return p;
    };
    float* GPt  = (float*)take(4096 * 4);
    float* JNt  = (float*)take(4096 * 4);
    float* refs = (float*)take(64);
    float* X    = (float*)take((size_t)N * 320 * 4);
    u16*   Xp   = (u16*)  take((size_t)N * 672 * 2);
    u16*   Wb   = (u16*)  take((size_t)672 * 640 * 2);
    float* T1   = (float*)take((size_t)N * 640 * 4);
    float* T2   = (float*)take((size_t)N * 640 * 4);
    float* T3   = (float*)take((size_t)N * 640 * 4);
    float* T4   = (float*)take((size_t)N * 640 * 4);
    u16*   Qf   = (u16*)  take((size_t)Bz * H * S * 32 * 2);
    u16*   Kf   = (u16*)  take((size_t)Bz * H * S * 32 * 2);
    u16*   Vf   = (u16*)  take((size_t)Bz * H * S * 48 * 2);

    auto A = [&](int i) { return (const float*)d_in[i]; };
    const float* in_mv = A(0);
    const float* in_sc = A(1);

    build_tables<<<1, 256, 0, stream>>>(GPt, JNt);
    refscale_k<<<Bz, 256, 0, stream>>>(in_mv, refs, S);

    auto prep = [&](int pi, int cin, int cout, int sin, int sout,
                    int Kp, int Mp, bool has_sout) {
        const float* wmv   = A(pi);
        const float* bmv   = A(pi + 1);
        const float* ws2mv = (sin > 0) ? A(pi + 2) : nullptr;
        const float* wss = nullptr, *wmv2s = nullptr, *bs = nullptr;
        if (has_sout) { wss = A(pi + 3); wmv2s = A(pi + 4); bs = A(pi + 5); }
        int total = Kp * Mp;
        prep_w<<<(total + 255) / 256, 256, 0, stream>>>(
            wmv, bmv, ws2mv, wss, wmv2s, bs, cin, cout, sin, sout, Kp, Mp, Wb);
    };
    auto gemm = [&](int Kp, int Mp, float* Out) {
        dim3 g(Mp / 64, N / 64);
        gemm_bf16<<<g, 256, 0, stream>>>(Xp, Wb, Out, N, Kp, Mp);
    };
    auto pack = [&](const float* mvp, int mvs, const float* sp, int ss,
                    int cin, int sin, int Kp, int norm) {
        pack_k<<<N / 8, 256, 0, stream>>>(mvp, mvs, sp, ss, cin, sin, Kp, norm, Xp, N);
    };

    // ---- lin_in : (4 mv,16 s) -> (16 mv, 64 s)
    pack(in_mv, 64, in_sc, 16, 4, 16, 96, 0);
    prep(2, 4, 16, 16, 64, 96, 320, true);
    gemm(96, 320, X);

    int pb = 8;  // first block's params
    for (int blk = 0; blk < 6; blk++) {
        int q = pb, k = pb + 6, v = pb + 12, ao = pb + 18;
        int lf = pb + 24, rt = pb + 30, mo = pb + 33;

        // ---- attention: norm -> q/k/v gemms -> flash attn -> out proj -> residual
        pack(X, 320, X + 256, 320, 16, 64, 352, 1);
        prep(q, 16, 16, 64, 64, 352, 320, true);  gemm(352, 320, T1);
        prep(k, 16, 16, 64, 64, 352, 320, true);  gemm(352, 320, T2);
        prep(v, 16, 16, 64, 64, 352, 320, true);  gemm(352, 320, T3);
        qkvprep_k<<<(N * 8 + 255) / 256, 256, 0, stream>>>(T1, T2, T3, Qf, Kf, Vf, N);
        attn_k<<<Bz * H * 16, 128, 0, stream>>>(Qf, Kf, Vf, T4, S);
        pack(T4, 320, T4 + 256, 320, 16, 64, 352, 0);
        prep(ao, 16, 16, 64, 64, 352, 320, true); gemm(352, 320, T1);
        add_k<<<(N * 320 + 255) / 256, 256, 0, stream>>>(X, T1, N * 320);

        // ---- geometric MLP: norm -> left/right gemms -> GP/JOIN gate -> out -> residual
        pack(X, 320, X + 256, 320, 16, 64, 352, 1);
        prep(lf, 16, 32, 64, 128, 352, 640, true); gemm(352, 640, T1);
        prep(rt, 16, 32, 64, 0,   352, 512, false); gemm(352, 512, T2);
        bilinear_k<<<N * 32 / 256, 256, 0, stream>>>(T1, T2, GPt, JNt, refs, T4, N);
        sgelu_k<<<(N * 128 + 255) / 256, 256, 0, stream>>>(T1, T4, N);
        pack(T4, 640, T4 + 512, 640, 32, 128, 672, 0);
        prep(mo, 32, 16, 128, 64, 672, 320, true); gemm(672, 320, T1);
        add_k<<<(N * 320 + 255) / 256, 256, 0, stream>>>(X, T1, N * 320);

        pb += 39;
    }

    // ---- lin_out : (16 mv,64 s) -> (4 mv,16 s)
    pack(X, 320, X + 256, 320, 16, 64, 352, 0);
    prep(pb, 16, 4, 64, 16, 352, 128, true);
    gemm(352, 128, T1);
    unpack_k<<<(N * 80 + 255) / 256, 256, 0, stream>>>(T1, (float*)d_out, N);
}
